// SlabSwinTransformer_65180423685307
// MI455X (gfx1250) — matrix operand order repaired and checked by reading, not compiled, not run on hardware
//
#include <hip/hip_runtime.h>
#include <hip/hip_bf16.h>

typedef _Float16 f16;
typedef _Float16 v16h __attribute__((ext_vector_type(16)));
typedef _Float16 v8h  __attribute__((ext_vector_type(8)));
typedef float    v8f  __attribute__((ext_vector_type(8)));

#define TOK   100352   // 8*112*112
#define NWIN  2048
#define CDIM  192
#define HID   768

enum { EPI_QKV = 0, EPI_PROJ = 1, EPI_GELU = 2, EPI_FC2 = 3 };
enum { MODE_WIN = 0, MODE_LIN = 1 };

static __device__ __forceinline__ v8h zero8() {
    v8h z;
    #pragma unroll
    for (int e = 0; e < 8; ++e) z[e] = (_Float16)0;
    return z;
}
static __device__ __forceinline__ v16h cat16(v8h lo, v8h hi) {
    return __builtin_shufflevector(lo, hi, 0,1,2,3,4,5,6,7,8,9,10,11,12,13,14,15);
}

// -------------------------------------------------------------------------
// f32 (K x N) -> f16 transposed (N x K): d[n*K+k] = s[k*N+n]
__global__ void cvtT_f32_f16(const float* __restrict__ s, f16* __restrict__ d,
                             int K, int N) {
    int i = blockIdx.x * 256 + threadIdx.x;
    if (i < N * K) {
        int n = i / K, k = i % K;
        d[i] = (f16)s[(long)k * N + n];
    }
}

// -------------------------------------------------------------------------
// LayerNorm (+ optional cyclic shift & window partition) -> f16 rows
__global__ __launch_bounds__(192)
void ln_kernel(const float* __restrict__ src, const float* __restrict__ g,
               const float* __restrict__ be, f16* __restrict__ dst, int mode) {
    __shared__ float red[2][6];
    const int t = blockIdx.x;          // token id (window-linear or linear)
    const int c = threadIdx.x;         // channel 0..191
    long srcIdx;
    if (mode == MODE_WIN) {
        int win = t / 49, n = t % 49;
        int b = win >> 8, wi = win & 255;
        int wh = wi >> 4, ww = wi & 15;
        int i = n / 7, j = n % 7;
        int sh = (wh * 7 + i + 3) % 112;   // roll(-3): xs[h] = xn[(h+3)%112]
        int sw = (ww * 7 + j + 3) % 112;
        srcIdx = ((long)b * 12544 + (long)sh * 112 + sw) * CDIM + c;
    } else {
        srcIdx = (long)t * CDIM + c;
    }
    float v = src[srcIdx];
    float s = v, s2 = v * v;
    #pragma unroll
    for (int off = 16; off; off >>= 1) {
        s  += __shfl_xor(s,  off);
        s2 += __shfl_xor(s2, off);
    }
    int wv = threadIdx.x >> 5, lane = threadIdx.x & 31;
    if (lane == 0) { red[0][wv] = s; red[1][wv] = s2; }
    __syncthreads();
    float ts = 0.f, ts2 = 0.f;
    #pragma unroll
    for (int k = 0; k < 6; ++k) { ts += red[0][k]; ts2 += red[1][k]; }
    float mu  = ts  * (1.0f / 192.0f);
    float var = ts2 * (1.0f / 192.0f) - mu * mu;
    float rs  = rsqrtf(var + 1e-5f);
    dst[(long)t * CDIM + c] = (f16)((v - mu) * rs * g[c] + be[c]);
}

// -------------------------------------------------------------------------
// Generic tiled WMMA GEMM.
//   A  : f16 [M x K] row-major (lda = K), M = 64-padded token block
//   Bt : f16 [N x K] row-major (weights pre-transposed) -> coalesced frags
// Epilogues: QKV (Q|K rows + transposed V), PROJ(+resid,unshift), GELU, FC2(+resid)
__global__ __launch_bounds__(128)
void gemm_kernel(const f16* __restrict__ A, const f16* __restrict__ Bt,
                 const float* __restrict__ bias, int K, int Ncols,
                 int mode, int epi,
                 f16* __restrict__ outH, float* __restrict__ outF,
                 const float* __restrict__ resid, f16* __restrict__ vtOut) {
    __shared__ f16 As[64][72];   // [m][k]
    __shared__ f16 Bs[64][72];   // [n][k]
    const int tid  = threadIdx.x;
    const int lane = tid & 31;
    const int wv   = tid >> 5;     // 0..3 : M-tile of this wave
    const int hi   = lane >> 4;
    const int lc   = lane & 15;

    long rowbase; int valid;
    if (mode == MODE_WIN) { rowbase = (long)blockIdx.x * 49; valid = 49; }
    else                  { rowbase = (long)blockIdx.x * 64; valid = 64; }

    const int nstrips = Ncols >> 6;
    const int kchunks = K >> 6;

    for (int ns = 0; ns < nstrips; ++ns) {
        v8f acc[4];
        #pragma unroll
        for (int t = 0; t < 4; ++t)
            #pragma unroll
            for (int r = 0; r < 8; ++r) acc[t][r] = 0.0f;

        for (int ko = 0; ko < kchunks; ++ko) {
            __syncthreads();
            // cooperative 64x64 chunk loads: A rows (m,k), Bt rows (n,k); all 16B
            #pragma unroll
            for (int i = 0; i < 4; ++i) {
                int cchunk = tid + i * 128;
                int row = cchunk >> 3;
                int cs  = (cchunk & 7) << 3;
                float4 av = make_float4(0.f, 0.f, 0.f, 0.f);
                if (row < valid)
                    av = *(const float4*)(A + (rowbase + row) * (long)K + (long)ko * 64 + cs);
                *(float4*)(&As[row][cs]) = av;
                float4 bv = *(const float4*)(Bt + ((long)(ns * 64 + row)) * K + (long)ko * 64 + cs);
                *(float4*)(&Bs[row][cs]) = bv;
            }
            if (ko + 1 < kchunks)
                __builtin_prefetch(A + (rowbase + (wv << 4) + lc) * (long)K + (long)(ko + 1) * 64, 0, 1);
            __syncthreads();

            #pragma unroll
            for (int kk = 0; kk < 2; ++kk) {
                // A frag: lane row fixed, k runs {8hi..}+{16+8hi..} -> 2 vector loads
                const f16* ap = &As[(wv << 4) + lc][(kk << 5) + (hi << 3)];
                v16h af = cat16(*(const v8h*)ap, *(const v8h*)(ap + 16));
                #pragma unroll
                for (int tj = 0; tj < 4; ++tj) {
                    // B frag: lane col fixed, k runs {8hi..}+{16+8hi..} -> 2 vector loads
                    const f16* bp = &Bs[(tj << 4) + lc][(kk << 5) + (hi << 3)];
                    v16h bf = cat16(*(const v8h*)bp, *(const v8h*)(bp + 16));
                    acc[tj] = __builtin_amdgcn_wmma_f32_16x16x32_f16(
                        false, af, false, bf, (short)0, acc[tj], false, false);
                }
            }
        }

        // ---- epilogue ----
        #pragma unroll
        for (int tj = 0; tj < 4; ++tj) {
            int colg = ns * 64 + (tj << 4) + lc;
            #pragma unroll
            for (int r = 0; r < 8; ++r) {
                int m = (wv << 4) + r + (hi << 3);
                float v = acc[tj][r] + (bias ? bias[colg] : 0.0f);
                long row = rowbase + m;
                if (epi == EPI_QKV) {
                    if (colg < 384) {
                        if (m < valid) outH[row * 384 + colg] = (f16)v;   // Q|K rows
                    } else {
                        int hd_ = colg - 384;
                        int head = hd_ >> 5, d = hd_ & 31;
                        long vidx = (((long)blockIdx.x * 6 + head) * 32 + d) * 64 + m;
                        vtOut[vidx] = (m < valid) ? (f16)v : (f16)0.0f;   // V^T, zero-padded
                    }
                } else if (m < valid) {
                    if (epi == EPI_GELU) {
                        float gl = 0.5f * v * (1.0f + erff(v * 0.70710678118654752f));
                        outH[row * (long)Ncols + colg] = (f16)gl;
                    } else if (epi == EPI_FC2) {
                        long idx = row * CDIM + colg;
                        outF[idx] = resid[idx] + v;
                    } else { // EPI_PROJ: window reverse + unshift + residual
                        int win = blockIdx.x;
                        int b = win >> 8, wi = win & 255;
                        int wh = wi >> 4, ww = wi & 15;
                        int i = m / 7, j = m % 7;
                        int oh = (wh * 7 + i + 3) % 112;  // x[(gh+3)%112] = xr[gh]
                        int ow = (ww * 7 + j + 3) % 112;
                        long dst = ((long)b * 12544 + (long)oh * 112 + ow) * CDIM + colg;
                        outF[dst] = resid[dst] + v;
                    }
                }
            }
        }
    }
}

// -------------------------------------------------------------------------
// Windowed attention: block = one window (2048 blocks), wave = one head.
//   qk : f16 [win*49+n][384]  (q | k, each 192 = 6 heads * 32)
//   vt : f16 [win][head][d(32)][m(64)]  zero-padded rows
__global__ __launch_bounds__(192)
void attn_kernel(const f16* __restrict__ qk, const f16* __restrict__ vt,
                 const float* __restrict__ rpb, f16* __restrict__ outA) {
    __shared__ f16 pbuf[6][64][72];
    const int win  = blockIdx.x;
    const int head = threadIdx.x >> 5;
    const int lane = threadIdx.x & 31;
    const int hi   = lane >> 4;
    const int lc   = lane & 15;
    const long base = (long)win * 49 * 384;

    // ---- Q fragments (A layout: 2 contiguous runs of 8 in d) ----
    v16h qf[4], kf[4];
    #pragma unroll
    for (int ti = 0; ti < 4; ++ti) {
        int m = (ti << 4) + lc;
        if (m < 49) {
            const f16* p = qk + base + (long)m * 384 + head * 32 + (hi << 3);
            qf[ti] = cat16(*(const v8h*)p, *(const v8h*)(p + 16));
        } else qf[ti] = cat16(zero8(), zero8());
    }
    // ---- K fragments (B layout: 2 contiguous runs of 8 in d) ----
    #pragma unroll
    for (int tj = 0; tj < 4; ++tj) {
        int m = (tj << 4) + lc;
        if (m < 49) {
            const f16* p = qk + base + (long)m * 384 + 192 + head * 32 + (hi << 3);
            kf[tj] = cat16(*(const v8h*)p, *(const v8h*)(p + 16));
        } else kf[tj] = cat16(zero8(), zero8());
    }

    // ---- S = Q K^T  (16 wmma, full K=32 each) ----
    v8f s[4][4];
    #pragma unroll
    for (int ti = 0; ti < 4; ++ti)
        #pragma unroll
        for (int tj = 0; tj < 4; ++tj) {
            v8f z;
            #pragma unroll
            for (int r = 0; r < 8; ++r) z[r] = 0.0f;
            s[ti][tj] = __builtin_amdgcn_wmma_f32_16x16x32_f16(
                false, qf[ti], false, kf[tj], (short)0, z, false, false);
        }

    // ---- scale + rel-pos bias + SW-MSA shift mask ----
    const int wi = win & 255;
    const int wh = wi >> 4, ww = wi & 15;
    const float scale = 0.17677669529663689f;  // 1/sqrt(32)
    #pragma unroll
    for (int ti = 0; ti < 4; ++ti)
        #pragma unroll
        for (int tj = 0; tj < 4; ++tj)
            #pragma unroll
            for (int r = 0; r < 8; ++r) {
                int nr = (ti << 4) + r + (hi << 3);
                int nc = (tj << 4) + lc;
                float v;
                if (nr < 49 && nc < 49) {
                    int i1 = nr / 7, j1 = nr % 7, i2 = nc / 7, j2 = nc % 7;
                    int idx = (i1 - i2 + 6) * 13 + (j1 - j2 + 6);
                    v = s[ti][tj][r] * scale + rpb[idx * 6 + head];
                    int gh1 = wh * 7 + i1, gw1 = ww * 7 + j1;
                    int gh2 = wh * 7 + i2, gw2 = ww * 7 + j2;
                    int r1 = (gh1 < 105 ? 0 : (gh1 < 109 ? 1 : 2)) * 3 + (gw1 < 105 ? 0 : (gw1 < 109 ? 1 : 2));
                    int r2 = (gh2 < 105 ? 0 : (gh2 < 109 ? 1 : 2)) * 3 + (gw2 < 105 ? 0 : (gw2 < 109 ? 1 : 2));
                    if (r1 != r2) v -= 100.0f;
                } else {
                    v = -1e30f;   // padded columns -> exp() = 0 exactly
                }
                s[ti][tj][r] = v;
            }

    // ---- softmax over 64 cols (cross-lane within each 16-lane half) ----
    #pragma unroll
    for (int ti = 0; ti < 4; ++ti) {
        #pragma unroll
        for (int r = 0; r < 8; ++r) {
            float mx = fmaxf(fmaxf(s[ti][0][r], s[ti][1][r]),
                             fmaxf(s[ti][2][r], s[ti][3][r]));
            #pragma unroll
            for (int off = 8; off; off >>= 1) mx = fmaxf(mx, __shfl_xor(mx, off));
            float ev[4], sm = 0.f;
            #pragma unroll
            for (int tj = 0; tj < 4; ++tj) { ev[tj] = __expf(s[ti][tj][r] - mx); sm += ev[tj]; }
            #pragma unroll
            for (int off = 8; off; off >>= 1) sm += __shfl_xor(sm, off);
            float inv = 1.0f / sm;
            int nr = (ti << 4) + r + (hi << 3);
            #pragma unroll
            for (int tj = 0; tj < 4; ++tj)
                pbuf[head][nr][(tj << 4) + lc] = (f16)(ev[tj] * inv);
        }
    }

    // ---- O = P V  (K = 64 padded: 2 chunks of 32; 16 wmma) ----
    v8f o[4][2];
    #pragma unroll
    for (int ti = 0; ti < 4; ++ti)
        #pragma unroll
        for (int dj = 0; dj < 2; ++dj)
            #pragma unroll
            for (int r = 0; r < 8; ++r) o[ti][dj][r] = 0.0f;

    #pragma unroll
    for (int kk = 0; kk < 2; ++kk) {
        v16h vfr[2];
        #pragma unroll
        for (int dj = 0; dj < 2; ++dj) {
            int d = (dj << 4) + lc;
            // V^T row (head,d): k = m runs {kk*32+8hi..}+{kk*32+16+8hi..}
            const f16* p = vt + (((long)win * 6 + head) * 32 + d) * 64 + (kk << 5) + (hi << 3);
            vfr[dj] = cat16(*(const v8h*)p, *(const v8h*)(p + 16));
        }
        #pragma unroll
        for (int ti = 0; ti < 4; ++ti) {
            const f16* pp = &pbuf[head][(ti << 4) + lc][(kk << 5) + (hi << 3)];
            v16h pf = cat16(*(const v8h*)pp, *(const v8h*)(pp + 16));
            #pragma unroll
            for (int dj = 0; dj < 2; ++dj)
                o[ti][dj] = __builtin_amdgcn_wmma_f32_16x16x32_f16(
                    false, pf, false, vfr[dj], (short)0, o[ti][dj], false, false);
        }
    }

    // ---- store per-head output: [win*49+n][head*32+d] f16 ----
    #pragma unroll
    for (int ti = 0; ti < 4; ++ti)
        #pragma unroll
        for (int dj = 0; dj < 2; ++dj)
            #pragma unroll
            for (int r = 0; r < 8; ++r) {
                int nr = (ti << 4) + r + (hi << 3);
                if (nr < 49) {
                    int d = (dj << 4) + lc;
                    outA[((long)win * 49 + nr) * CDIM + head * 32 + d] = (f16)o[ti][dj][r];
                }
            }
}

// -------------------------------------------------------------------------
extern "C" void kernel_launch(void* const* d_in, const int* in_sizes, int n_in,
                              void* d_out, int out_size, void* d_ws, size_t ws_size,
                              hipStream_t stream) {
    const float* x      = (const float*)d_in[0];
    const float* n1g    = (const float*)d_in[1];
    const float* n1b    = (const float*)d_in[2];
    const float* qkv_w  = (const float*)d_in[3];
    const float* qkv_b  = (const float*)d_in[4];
    const float* rpb    = (const float*)d_in[5];
    const float* proj_w = (const float*)d_in[6];
    const float* proj_b = (const float*)d_in[7];
    const float* n2g    = (const float*)d_in[8];
    const float* n2b    = (const float*)d_in[9];
    const float* fc1_w  = (const float*)d_in[10];
    const float* fc1_b  = (const float*)d_in[11];
    const float* fc2_w  = (const float*)d_in[12];
    const float* fc2_b  = (const float*)d_in[13];

    char* ws = (char*)d_ws;
    size_t off = 0;
    auto alloc = [&](size_t bytes) -> void* {
        void* p = ws + off;
        off = (off + bytes + 255) & ~(size_t)255;
        return p;
    };
    f16* qkvWt  = (f16*)alloc((size_t)576 * CDIM * 2);   // [576][192]
    f16* projWt = (f16*)alloc((size_t)CDIM * CDIM * 2);  // [192][192]
    f16* fc1Wt  = (f16*)alloc((size_t)HID * CDIM * 2);   // [768][192]
    f16* fc2Wt  = (f16*)alloc((size_t)CDIM * HID * 2);   // [192][768]
    f16* R1     = (f16*)alloc((size_t)TOK * CDIM * 2);   // win_h / attn_h / x2_h
    f16* R2     = (f16*)alloc((size_t)TOK * HID * 2);    // {QK + V^T} then fc1-out
    float* Y    = (float*)alloc((size_t)TOK * CDIM * 4); // x + attn branch

    f16* QK = R2;                                     // [TOK][384]
    f16* VT = R2 + (size_t)TOK * 384;                 // [NWIN][6][32][64]
    f16* Hh = R2;                                     // [TOK][768] (after attn)

    // 0) weights -> f16, transposed to [N][K]
    cvtT_f32_f16<<<(576 * CDIM + 255) / 256, 256, 0, stream>>>(qkv_w, qkvWt, CDIM, 576);
    cvtT_f32_f16<<<(CDIM * CDIM + 255) / 256, 256, 0, stream>>>(proj_w, projWt, CDIM, CDIM);
    cvtT_f32_f16<<<(HID * CDIM + 255) / 256, 256, 0, stream>>>(fc1_w, fc1Wt, CDIM, HID);
    cvtT_f32_f16<<<(CDIM * HID + 255) / 256, 256, 0, stream>>>(fc2_w, fc2Wt, HID, CDIM);

    // 1) LN1 + shift + window partition -> f16 window-linear rows
    ln_kernel<<<TOK, 192, 0, stream>>>(x, n1g, n1b, R1, MODE_WIN);

    // 2) QKV GEMM: (100352 x 192) x (192 x 576) -> Q|K rows + transposed V
    gemm_kernel<<<NWIN, 128, 0, stream>>>(R1, qkvWt, qkv_b, CDIM, 576,
                                          MODE_WIN, EPI_QKV, QK, nullptr, nullptr, VT);

    // 3) windowed attention (bias + shift-mask + softmax, all WMMA)
    attn_kernel<<<NWIN, 192, 0, stream>>>(QK, VT, rpb, R1);

    // 4) proj GEMM + window-reverse + unshift + residual -> Y (f32)
    gemm_kernel<<<NWIN, 128, 0, stream>>>(R1, projWt, proj_b, CDIM, CDIM,
                                          MODE_WIN, EPI_PROJ, nullptr, Y, x, nullptr);

    // 5) LN2 on Y -> f16 linear rows
    ln_kernel<<<TOK, 192, 0, stream>>>(Y, n2g, n2b, R1, MODE_LIN);

    // 6) FC1 GEMM + exact GELU -> f16
    gemm_kernel<<<TOK / 64, 128, 0, stream>>>(R1, fc1Wt, fc1_b, CDIM, HID,
                                              MODE_LIN, EPI_GELU, Hh, nullptr, nullptr, nullptr);

    // 7) FC2 GEMM + residual(Y) -> d_out (f32)
    gemm_kernel<<<TOK / 64, 128, 0, stream>>>(Hh, fc2Wt, fc2_b, HID, CDIM,
                                              MODE_LIN, EPI_FC2, nullptr, (float*)d_out, Y, nullptr);
}
